// ConcatSquashLinearSA_69020124446957
// MI455X (gfx1250) — matrix operand order repaired and checked by reading, not compiled, not run on hardware
//
#include <hip/hip_runtime.h>
#include <hip/hip_bf16.h>

// ---------------------------------------------------------------------------
// Problem constants (match reference)
// ---------------------------------------------------------------------------
#define BB       16
#define NN       16384
#define DIM      256         // DIM_IN == DIM_OUT == 256
#define DCTX     131
#define NWGB     16          // workgroups per batch in main GEMM
#define ROWS_WG  (NN / NWGB) // 1024 rows per workgroup
#define LDS_STR  264         // bf16 row stride in LDS (256 + 8 pad) = 528 B, 16B-aligned

typedef __bf16       v16bf __attribute__((ext_vector_type(16)));
typedef float        v8f   __attribute__((ext_vector_type(8)));
typedef float        v4f   __attribute__((ext_vector_type(4)));
typedef unsigned int v4u   __attribute__((ext_vector_type(4)));
typedef unsigned int u32x4 __attribute__((ext_vector_type(4)));
typedef int          i32x8 __attribute__((ext_vector_type(8)));
typedef int          i32x4 __attribute__((ext_vector_type(4)));

union BF16x16 {
    v16bf v;
    __bf16 h[16];
    unsigned short u[16];
    v4u q[2];
};

__device__ __forceinline__ unsigned short f2bf(float f) {
    unsigned u = __builtin_bit_cast(unsigned, f);
    unsigned r = u + 0x7FFFu + ((u >> 16) & 1u);   // round-to-nearest-even
    return (unsigned short)(r >> 16);
}

#if __has_builtin(__builtin_amdgcn_tensor_load_to_lds) && \
    __has_builtin(__builtin_amdgcn_s_wait_tensorcnt)
#define USE_TDM 1
#endif
#if __has_include(<hip/amd_detail/amd_gfx1250_TDM.h>)
#define TDM_ARITY6 1
#endif

// ---------------------------------------------------------------------------
// Setup kernel 1: per batch, compute gate / bias / kv, softmax'd + col-norm'd
// attention with identity added: A_full = I + att2  ->  ws (f32, 16x256x256)
// ---------------------------------------------------------------------------
__global__ __launch_bounds__(256) void setup_attention(
    const float* __restrict__ ctx,    const float* __restrict__ W_gate,
    const float* __restrict__ b_gate, const float* __restrict__ W_bias,
    const float* __restrict__ W_kv,
    float* __restrict__ A_full, float* __restrict__ gate_ws,
    float* __restrict__ bias_ws)
{
    const int b = blockIdx.x;          // 0..15
    const int e = threadIdx.x;         // 0..255
    __shared__ float skv[DIM], srm[DIM], srs[DIM];

    const float* c = ctx + b * DCTX;
    float g = b_gate[e], bi = 0.f, kv = 0.f;
    for (int i = 0; i < DCTX; ++i) {
        float cv = c[i];
        g  = fmaf(cv, W_gate[e * DCTX + i], g);
        bi = fmaf(cv, W_bias[e * DCTX + i], bi);
        kv = fmaf(cv, W_kv  [e * DCTX + i], kv);
    }
    float ge = 1.f / (1.f + __expf(-g));
    gate_ws[b * DIM + e] = ge;
    bias_ws[b * DIM + e] = bi;
    skv[e] = kv;
    __syncthreads();

    float mx = skv[0], mn = skv[0];
    for (int i = 1; i < DIM; ++i) { float t = skv[i]; mx = fmaxf(mx, t); mn = fminf(mn, t); }
    float rm = (kv >= 0.f) ? kv * mx : kv * mn;
    float rs = 0.f;
    for (int i = 0; i < DIM; ++i) rs += __expf(fmaf(kv, skv[i], -rm));
    srm[e] = rm; srs[e] = rs;      // indexed by row o == e here
    __syncthreads();

    float cs = 0.f;
    for (int o = 0; o < DIM; ++o) cs += __expf(fmaf(skv[o], kv, -srm[o])) / srs[o];
    float inv_cs = 1.f / (1e-9f + cs);

    float* Ab = A_full + ((size_t)b * DIM) * DIM;
    for (int o = 0; o < DIM; ++o) {
        float a = __expf(fmaf(skv[o], kv, -srm[o])) / srs[o] * inv_cs;
        if (o == e) a += 1.f;
        Ab[o * DIM + e] = a;       // coalesced across e
    }
}

// ---------------------------------------------------------------------------
// Setup kernel 2: fold W_layer, gate into Mg (bf16, TRANSPOSED: [b][e][d]),
// and fold b_layer, gate, bias into v vector.  One block per (b,e) column.
// ---------------------------------------------------------------------------
__global__ __launch_bounds__(256) void setup_fold(
    const float* __restrict__ W_layer, const float* __restrict__ b_layer,
    const float* __restrict__ A_full,  const float* __restrict__ gate_ws,
    const float* __restrict__ bias_ws,
    unsigned short* __restrict__ Mt, float* __restrict__ vvec)
{
    const int e = blockIdx.x & (DIM - 1);
    const int b = blockIdx.x >> 8;
    const int d = threadIdx.x;
    const float* Ab = A_full + ((size_t)b * DIM) * DIM;

    float m = 0.f;
    for (int o = 0; o < DIM; ++o)
        m = fmaf(W_layer[o * DIM + d], Ab[o * DIM + e], m);

    const float ge = gate_ws[b * DIM + e];
    Mt[((size_t)b * DIM + e) * DIM + d] = f2bf(m * ge);

    __shared__ float red[DIM];
    red[d] = b_layer[d] * Ab[d * DIM + e];
    __syncthreads();
    for (int s = 128; s > 0; s >>= 1) {
        if (d < s) red[d] += red[d + s];
        __syncthreads();
    }
    if (d == 0) vvec[b * DIM + e] = red[0] * ge + bias_ws[b * DIM + e];
}

// ---------------------------------------------------------------------------
// A-fragment loader (nontemporal: x is streamed exactly once).
// ISA 16-bit 16x32 layout:
//   lane(hi=0): v0-3 = K[kb+0..7],  v4-7 = K[kb+16..23]
//   lane(hi=1): v0-3 = K[kb+8..15], v4-7 = K[kb+24..31]
// ---------------------------------------------------------------------------
__device__ __forceinline__ BF16x16 load_afrag(const float* xp) {
    v4f f0 = __builtin_nontemporal_load((const v4f*)(xp));
    v4f f1 = __builtin_nontemporal_load((const v4f*)(xp + 4));
    v4f f2 = __builtin_nontemporal_load((const v4f*)(xp + 16));
    v4f f3 = __builtin_nontemporal_load((const v4f*)(xp + 20));
    BF16x16 af;
#pragma unroll
    for (int j = 0; j < 4; ++j) {
        af.h[j]      = (__bf16)f0[j];
        af.h[4 + j]  = (__bf16)f1[j];
        af.h[8 + j]  = (__bf16)f2[j];
        af.h[12 + j] = (__bf16)f3[j];
    }
    return af;
}

// ---------------------------------------------------------------------------
// Main GEMM: out[b,n,e] = sum_d x[b,n,d] * Mg_b[d,e] + v_b[e]
// One WG per (batch, row-chunk). Mg_b^T staged in LDS (bf16, padded) by TDM.
// Each wave computes 32x256 super-strips: each B fragment feeds TWO WMMAs.
// (No manual B pipelining: a single B buffer is what keeps this spill-free.)
// ---------------------------------------------------------------------------
__global__ __launch_bounds__(256) void fused_gemm(
    const float* __restrict__ x, const unsigned short* __restrict__ Mt,
    const float* __restrict__ vvec, float* __restrict__ out)
{
    extern __shared__ unsigned char smem[];
    unsigned short* sM = (unsigned short*)smem;                       // [256][LDS_STR]
    float*          sV = (float*)(smem + DIM * LDS_STR * sizeof(unsigned short));

    const int tid  = threadIdx.x;
    const int b    = blockIdx.x / NWGB;
    const int chnk = blockIdx.x % NWGB;
    const unsigned short* gM = Mt + ((size_t)b * DIM) * DIM;

#if defined(USE_TDM)
    // ---- Tensor Data Mover: one instruction stages the full 256x256 bf16
    //      tile, HW-padding 4 dwords every 128 dwords -> 528B LDS row stride.
    if (tid < 32) {
        unsigned long long ga = (unsigned long long)(uintptr_t)gM;
        u32x4 g0;
        g0[0] = 1u;                                            // count=1 (valid user D#)
        g0[1] = __builtin_amdgcn_groupstaticsize();            // lds_addr (dyn LDS base)
        g0[2] = (unsigned)(ga & 0xFFFFFFFFu);                  // global_addr[31:0]
        g0[3] = (unsigned)((ga >> 32) & 0x01FFFFFFu) | (2u << 30); // addr[56:32] | type=2
        i32x8 g1;
        g1[0] = (1 << 16)    // data_size = 2 bytes
              | (1 << 20)    // pad_enable
              | (6 << 22)    // pad_interval: 128 dwords (one 512B row)
              | (3 << 25);   // pad_amount: 4 dwords (16B)
        g1[1] = (DIM & 0xFFFF) << 16;                          // tensor_dim0 lo16
        g1[2] = (DIM >> 16) | ((DIM & 0xFFFF) << 16);          // dim0 hi | dim1 lo16
        g1[3] = (DIM >> 16) | (DIM << 16);                     // dim1 hi | tile_dim0
        g1[4] = DIM;                                           // tile_dim1 (tile_dim2=0)
        g1[5] = DIM;                                           // tensor_dim0_stride lo32
        g1[6] = 0;
        g1[7] = 0;
        i32x4 z4 = (i32x4){0, 0, 0, 0};
#if defined(TDM_ARITY6)
        i32x8 z8 = (i32x8){0, 0, 0, 0, 0, 0, 0, 0};
        __builtin_amdgcn_tensor_load_to_lds(g0, g1, z4, z4, z8, 0);
#else
        __builtin_amdgcn_tensor_load_to_lds(g0, g1, z4, z4, 0);
#endif
        __builtin_amdgcn_s_wait_tensorcnt(0);
    }
#else
    // ---- fallback: plain vectorized copy ----
    for (int c = tid; c < DIM * 32; c += 256) {
        int e = c >> 5, j = c & 31;
        v4u val = ((const v4u*)(gM + (size_t)e * DIM))[j];
        *(v4u*)(&sM[e * LDS_STR + j * 8]) = val;
    }
#endif
    sV[tid] = vvec[b * DIM + tid];
    __syncthreads();

    const int wave = tid >> 5;
    const int lane = tid & 31;
    const int l2   = lane & 15;
    const int hi   = lane >> 4;

    const int wave_row0 = chnk * ROWS_WG + wave * (ROWS_WG / 8);      // 128 rows / wave

#pragma unroll 1
    for (int s = 0; s < 4; ++s) {                                     // 4 x 32-row strips
        const int row0 = wave_row0 + s * 32;
        const float* xrow0 = x + ((size_t)b * NN + row0 + l2) * DIM;
        const float* xrow1 = xrow0 + (size_t)16 * DIM;

        v8f acc0[16], acc1[16];
#pragma unroll
        for (int t = 0; t < 16; ++t) {
            acc0[t] = (v8f){0.f,0.f,0.f,0.f,0.f,0.f,0.f,0.f};
            acc1[t] = (v8f){0.f,0.f,0.f,0.f,0.f,0.f,0.f,0.f};
        }

#pragma unroll
        for (int k = 0; k < 8; ++k) {              // kbase = k*32
            BF16x16 a0 = load_afrag(xrow0 + k * 32 + hi * 8);
            BF16x16 a1 = load_afrag(xrow1 + k * 32 + hi * 8);
#pragma unroll
            for (int t = 0; t < 16; ++t) {
                // B fragment: lane col = l2, K = kb + hi*8 + 0..7, kb + 16 + hi*8 + 0..7
                BF16x16 bf;
                const v4u* pb = (const v4u*)(&sM[(t * 16 + l2) * LDS_STR + k * 32 + hi * 8]);
                bf.q[0] = pb[0];
                bf.q[1] = pb[2];
                acc0[t] = __builtin_amdgcn_wmma_f32_16x16x32_bf16(
                    false, a0.v, false, bf.v, (short)0, acc0[t], false, false);
                acc1[t] = __builtin_amdgcn_wmma_f32_16x16x32_bf16(
                    false, a1.v, false, bf.v, (short)0, acc1[t], false, false);
            }
        }

        // ---- epilogue: D layout -> row = i + hi*8, col = t*16 + l2 ----
        float* obase = out + ((size_t)b * NN + row0) * DIM;
#pragma unroll
        for (int t = 0; t < 16; ++t) {
            float vbv = sV[t * 16 + l2];
            float* ocol = obase + t * 16 + l2;
#pragma unroll
            for (int i = 0; i < 8; ++i) {
                __builtin_nontemporal_store(acc0[t][i] + vbv,
                                            &ocol[(size_t)(i + hi * 8) * DIM]);
                __builtin_nontemporal_store(acc1[t][i] + vbv,
                                            &ocol[(size_t)(16 + i + hi * 8) * DIM]);
            }
        }
    }
}

// ---------------------------------------------------------------------------
// Launch
// ---------------------------------------------------------------------------
extern "C" void kernel_launch(void* const* d_in, const int* in_sizes, int n_in,
                              void* d_out, int out_size, void* d_ws, size_t ws_size,
                              hipStream_t stream) {
    const float* ctx     = (const float*)d_in[0];
    const float* x       = (const float*)d_in[1];
    const float* W_layer = (const float*)d_in[2];
    const float* b_layer = (const float*)d_in[3];
    const float* W_bias  = (const float*)d_in[4];
    const float* W_gate  = (const float*)d_in[5];
    const float* b_gate  = (const float*)d_in[6];
    const float* W_kv    = (const float*)d_in[7];
    float* out = (float*)d_out;

    // workspace carve-up (~6.3 MB)
    float* A_full  = (float*)d_ws;                           // 16*256*256 f32
    float* gate_ws = A_full + (size_t)BB * DIM * DIM;        // 16*256 f32
    float* bias_ws = gate_ws + BB * DIM;                     // 16*256 f32
    float* vvec    = bias_ws + BB * DIM;                     // 16*256 f32
    unsigned short* Mt = (unsigned short*)(vvec + BB * DIM); // 16*256*256 bf16

    setup_attention<<<BB, 256, 0, stream>>>(ctx, W_gate, b_gate, W_bias, W_kv,
                                            A_full, gate_ws, bias_ws);
    setup_fold<<<BB * DIM, 256, 0, stream>>>(W_layer, b_layer, A_full,
                                             gate_ws, bias_ws, Mt, vvec);

    const size_t smem_bytes = (size_t)DIM * LDS_STR * sizeof(unsigned short)
                            + DIM * sizeof(float);           // 136192 B (CDNA5 WGP LDS)
    fused_gemm<<<BB * NWGB, 256, smem_bytes, stream>>>(x, Mt, vvec, out);
}
